// ClassBalance_34497177321947
// MI455X (gfx1250) — compile-verified, matrix operand order checked by reading, not run on hardware
//
#include <hip/hip_runtime.h>
#include <hip/hip_bf16.h>
#include <math.h>

typedef __attribute__((ext_vector_type(8))) int   v8i;
typedef __attribute__((ext_vector_type(4))) float v4f;

// ---------------------------------------------------------------------------
// Kernel 1: zero the 8-bin global histogram in workspace (graph-safe).
// ---------------------------------------------------------------------------
__global__ void __launch_bounds__(32) zero_hist_kernel(unsigned* __restrict__ ghist) {
    if (threadIdx.x < 8u) ghist[threadIdx.x] = 0u;
}

// ---------------------------------------------------------------------------
// Kernel 2: streaming argmax + WMMA-based histogram.
//   masks: [B, C=8, H, W] f32, channel stride = H*W = 1<<hwShift elements.
//   Each thread owns 4 CONSECUTIVE pixels per iteration and loads each
//   channel as one b128 NT load (8 x global_load_b128 per iteration).
//   The 4 argmax classes become 8-byte one-hots in an IU8 A-fragment;
//   B-fragment is the constant pattern B[k][n] = ((k&7)==n) for n<8.
//   D = A*B accumulates per-class counts in the i32 16x16 accumulator.
// ---------------------------------------------------------------------------
__global__ void __launch_bounds__(256) class_hist_wmma_kernel(
    const float* __restrict__ masks,
    unsigned* __restrict__ ghist,
    unsigned P,        // total pixels = B*H*W
    unsigned hwShift,  // log2(H*W)
    unsigned niter,    // WMMA iterations per thread (4 pixels each)
    unsigned T)        // total threads in grid
{
    __shared__ unsigned lhist[8];
    if (threadIdx.x < 8u) lhist[threadIdx.x] = 0u;
    __syncthreads();

    const unsigned gtid   = blockIdx.x * blockDim.x + threadIdx.x;
    const unsigned lane   = threadIdx.x & 31u;
    const unsigned hwMask = (1u << hwShift) - 1u;

    // Constant B fragment. For VGPR v of the B operand, byte b maps to
    // k with (k & 7) == (v&1)*4 + b (identical for lanes 0-15 and 16-31,
    // since the +16/+32 K offsets are multiples of 8). Column n = lane&15.
    const unsigned n  = lane & 15u;
    const unsigned we = (n < 4u) ? (1u << (8u * n)) : 0u;                   // even VGPRs
    const unsigned wo = (n >= 4u && n < 8u) ? (1u << (8u * (n - 4u))) : 0u; // odd VGPRs
    v8i Bf;
    Bf[0] = (int)we; Bf[1] = (int)wo; Bf[2] = (int)we; Bf[3] = (int)wo;
    Bf[4] = (int)we; Bf[5] = (int)wo; Bf[6] = (int)we; Bf[7] = (int)wo;

    v8i acc = {0, 0, 0, 0, 0, 0, 0, 0};

    for (unsigned it = 0; it < niter; ++it) {
        // Base pixel of this thread's 4-pixel group (multiple of 4, so the
        // group never crosses a channel boundary since H*W is a power of 2).
        const unsigned pb  = (it * T + gtid) * 4u;
        const unsigned b   = pb >> hwShift;
        const unsigned rem = pb & hwMask;
        const float* p = masks + (((size_t)(b << 3)) << hwShift) + rem;

        // argmax over 8 channels for 4 pixels at once (strict > keeps the
        // first occurrence of the max, matching jnp.argmax).
        float    bv[4];
        unsigned best[4];
        {
            const v4f v0 = __builtin_nontemporal_load((const v4f*)p);
            #pragma unroll
            for (unsigned i = 0; i < 4u; ++i) { bv[i] = v0[i]; best[i] = 0u; }
        }
        #pragma unroll
        for (unsigned c = 1u; c < 8u; ++c) {
            const v4f v = __builtin_nontemporal_load(
                (const v4f*)(p + ((size_t)c << hwShift)));
            #pragma unroll
            for (unsigned i = 0; i < 4u; ++i) {
                const bool g = v[i] > bv[i];
                bv[i]   = g ? v[i] : bv[i];
                best[i] = g ? c : best[i];
            }
        }

        // Pack 4 one-hots (8 bytes each) into the A fragment.
        v8i A;
        #pragma unroll
        for (unsigned i = 0; i < 4u; ++i) {
            const unsigned long long oh = 1ull << (best[i] * 8u);
            A[2u * i]      = (int)(unsigned)(oh & 0xffffffffull);
            A[2u * i + 1u] = (int)(unsigned)(oh >> 32);
        }

        // D[m][n] += sum_k A[m][k] * B[k][n]  -> per-class pixel counts
        acc = __builtin_amdgcn_wmma_i32_16x16x64_iu8(false, A, false, Bf, acc,
                                                     false, false);
    }

    // Reduce accumulator: sum 8 row-VGPRs, then add partner lane (^16);
    // both halves hold column n = lane&15.
    int partial = acc[0] + acc[1] + acc[2] + acc[3] +
                  acc[4] + acc[5] + acc[6] + acc[7];
    partial += __shfl_xor(partial, 16, 32);
    if (lane < 8u) atomicAdd(&lhist[lane], (unsigned)partial);

    // Tail pixels (if grid doesn't tile P exactly) — scalar path, no WMMA.
    for (unsigned q = niter * 4u * T + gtid; q < P; q += T) {
        const unsigned b   = q >> hwShift;
        const unsigned rem = q & hwMask;
        const float* p = masks + (((size_t)(b << 3)) << hwShift) + rem;
        float    bv   = __builtin_nontemporal_load(p);
        unsigned best = 0u;
        #pragma unroll
        for (unsigned c = 1u; c < 8u; ++c) {
            const float v = __builtin_nontemporal_load(p + ((size_t)c << hwShift));
            const bool g = v > bv;
            bv   = g ? v : bv;
            best = g ? c : best;
        }
        atomicAdd(&lhist[best], 1u);
    }

    __syncthreads();
    if (threadIdx.x < 8u) {
        atomicAdd(&ghist[threadIdx.x], lhist[threadIdx.x]);
    }
}

// ---------------------------------------------------------------------------
// Kernel 3: EMA update + balance norm. One wave.
//   out[0]   = || (dist - 1/C) / (1 - 1/C) ||_2
//   out[1..8]= dist
// ---------------------------------------------------------------------------
__global__ void __launch_bounds__(32) finalize_kernel(
    const unsigned* __restrict__ ghist,
    const float* __restrict__ prev,
    float* __restrict__ out,
    float invTotal)
{
    const unsigned lane = threadIdx.x;
    float dist = 0.0f, sq = 0.0f;
    if (lane < 8u) {
        const float h = (float)ghist[lane];
        dist = prev[lane] * 0.99f + 0.01f * h * invTotal;
        const float diff = (dist - 0.125f) / (1.0f - 0.125f);
        sq = diff * diff;
    }
    // sum of squares across lanes 0..7 (lanes >=8 contribute 0)
    sq += __shfl_xor(sq, 4, 32);
    sq += __shfl_xor(sq, 2, 32);
    sq += __shfl_xor(sq, 1, 32);
    if (lane == 0u) out[0] = sqrtf(sq);
    if (lane < 8u)  out[1u + lane] = dist;
}

// ---------------------------------------------------------------------------
extern "C" void kernel_launch(void* const* d_in, const int* in_sizes, int n_in,
                              void* d_out, int out_size, void* d_ws, size_t ws_size,
                              hipStream_t stream) {
    const float* masks = (const float*)d_in[0];  // [B, 8, 512, 512] f32
    const float* prev  = (const float*)d_in[1];  // [8] f32
    float*    out   = (float*)d_out;             // [1 + 8] f32
    unsigned* ghist = (unsigned*)d_ws;           // 8 x u32 scratch

    const unsigned P  = (unsigned)(in_sizes[0] / (int)in_sizes[1]); // B*H*W
    const unsigned hwShift = 18u;                // log2(512*512)
    (void)n_in; (void)out_size; (void)ws_size;

    const unsigned blocks  = 1024u;
    const unsigned threads = 256u;
    const unsigned T       = blocks * threads;   // 262144
    const unsigned niter   = P / (4u * T);       // 4 for the reference shape

    zero_hist_kernel<<<1, 32, 0, stream>>>(ghist);
    class_hist_wmma_kernel<<<blocks, threads, 0, stream>>>(
        masks, ghist, P, hwShift, niter, T);
    finalize_kernel<<<1, 32, 0, stream>>>(ghist, prev, out, 1.0f / (float)P);
}
